// CoinMRF_25941602467911
// MI455X (gfx1250) — compile-verified, matrix operand order checked by reading, not run on hardware
//
#include <hip/hip_runtime.h>
#include <math.h>

typedef __attribute__((ext_vector_type(8))) int v8i;

#define N_SAMPLES 65536
#define NUM_F 512
#define FT_COUNT 32                       // 512 / 16 feature tiles
#define TILES_PER_WAVE 32                 // 32 tiles x 64 samples = 2048 samples per wave
#define SAMPLES_PER_WAVE (TILES_PER_WAVE * 64)
#define SC_COUNT (N_SAMPLES / SAMPLES_PER_WAVE)   // 32 sample chunks
#define NUM_WAVES (FT_COUNT * SC_COUNT)           // 1024 waves
#define WAVES_PER_BLOCK 8
#define NUM_BLOCKS (NUM_WAVES / WAVES_PER_BLOCK)  // 128 blocks

// ---------------------------------------------------------------------------
// Kernel 1: precompute u[f] = (c+s)/2, v[f] = (c-s)/2 where
//           c = theta+psi, s = softplus(theta).
// Then per feature:  count1*c + countm1*s = t*u + d*v
// with t = count1+countm1 = sum|y|,  d = count1-countm1 = sum y.
// ---------------------------------------------------------------------------
__global__ void mnar_init(const float* __restrict__ theta,
                          const float* __restrict__ psi,
                          float* __restrict__ ws) {
    int f = threadIdx.x;
    if (f < NUM_F) {
        float th = theta[f];
        float c  = th + psi[f];
        float s  = fmaxf(th, 0.0f) + log1pf(expf(-fabsf(th)));  // stable softplus
        ws[f]         = 0.5f * (c + s);   // u
        ws[NUM_F + f] = 0.5f * (c - s);   // v
    }
}

// ---------------------------------------------------------------------------
// Kernel 2: accumulate t[f] = sum|y| and d[f] = sum y on the matrix pipe.
// Each wave: one 16-feature tile x 2048 samples; per 64-sample tile:
//   - pack low bytes of 4 int32 y's into one B VGPR via v_perm_b32 (3 perms)
//   - B_abs = B_signed & 0x01010101 (|y| since 0xFF&1 = 1)
//   - v_wmma_i32_16x16x64_iu8 (unsigned) for t, (signed, neg_lo) for d
// B layout (8-bit, 64x16, wave32): column j = lane & 15;
//   VGPRs 0..3: lanes 0-15 K=0-15,  lanes 16-31 K=16-31
//   VGPRs 4..7: lanes 0-15 K=32-47, lanes 16-31 K=48-63
// ---------------------------------------------------------------------------
__global__ __launch_bounds__(256) void mnar_count(const int* __restrict__ Y,
                                                  const float* __restrict__ ws,
                                                  float* __restrict__ partials) {
    const int lane = threadIdx.x & 31;
    const int w    = blockIdx.x * WAVES_PER_BLOCK + (threadIdx.x >> 5);
    const int ft   = w & (FT_COUNT - 1);   // feature tile   (FT_COUNT == 32)
    const int sc   = w >> 5;               // sample chunk
    const int f0   = ft * 16;
    const int n0   = sc * SAMPLES_PER_WAVE;

    const int j = lane & 15;               // column (feature within tile)
    const int g = lane >> 4;               // lane-group -> K offset of 16

    // A = all-ones bytes (valid as u8 +1 and as i8 +1; layout-invariant)
    v8i a_ones;
    #pragma unroll
    for (int i = 0; i < 8; ++i) a_ones[i] = 0x01010101;

    v8i acc_t = {};   // per-feature sum of |y|  (unsigned iu8)
    v8i acc_d = {};   // per-feature sum of  y   (signed   iu8)

    const int* yb = Y + (size_t)n0 * NUM_F + f0 + j;

    #pragma unroll 1
    for (int t = 0; t < TILES_PER_WAVE; ++t) {
        const int* ytile = yb + (size_t)t * 64 * NUM_F;

        v8i b_sgn, b_abs;
        #pragma unroll
        for (int v = 0; v < 8; ++v) {
            // K base for this VGPR per the 8-bit B layout
            const int kb = g * 16 + ((v >> 2) << 5) + ((v & 3) << 2);
            unsigned y0 = (unsigned)ytile[(size_t)(kb + 0) * NUM_F];
            unsigned y1 = (unsigned)ytile[(size_t)(kb + 1) * NUM_F];
            unsigned y2 = (unsigned)ytile[(size_t)(kb + 2) * NUM_F];
            unsigned y3 = (unsigned)ytile[(size_t)(kb + 3) * NUM_F];
            // pack low bytes: dst = {y3.b0, y2.b0, y1.b0, y0.b0}
            // sel k: 0..3 -> S1.byte[k], 4..7 -> S0.byte[k-4], 0x0C -> 0x00
            unsigned lo = __builtin_amdgcn_perm(y1, y0, 0x0C0C0400u); // b0=y0,b1=y1
            unsigned hi = __builtin_amdgcn_perm(y3, y2, 0x0C0C0400u); // b0=y2,b1=y3
            unsigned pk = __builtin_amdgcn_perm(hi, lo, 0x05040100u); // y0,y1,y2,y3
            b_sgn[v] = (int)pk;                     // i8 bytes: -1/0/+1
            b_abs[v] = (int)(pk & 0x01010101u);     // u8 bytes: |y|
        }

        acc_t = __builtin_amdgcn_wmma_i32_16x16x64_iu8(false, a_ones, false, b_abs,
                                                       acc_t, false, false);
        acc_d = __builtin_amdgcn_wmma_i32_16x16x64_iu8(true,  a_ones, true,  b_sgn,
                                                       acc_d, false, false);
    }

    // D row 0 lives in element 0: lanes 0-15 hold N=lane (row M=0),
    // lanes 16-31 hold N=lane-16 (row M=8) -> same value (A all-ones),
    // so mask the upper half to avoid double counting.
    const float uf = ws[f0 + j];
    const float vf = ws[NUM_F + f0 + j];
    float contrib = (g == 0)
        ? ((float)acc_t[0] * uf + (float)acc_d[0] * vf)
        : 0.0f;

    #pragma unroll
    for (int off = 16; off > 0; off >>= 1)
        contrib += __shfl_xor(contrib, off, 32);

    if (lane == 0) partials[w] = contrib;
}

// ---------------------------------------------------------------------------
// Kernel 3: deterministic tree reduction of wave partials -> -total/N
// ---------------------------------------------------------------------------
__global__ void mnar_finalize(const float* __restrict__ partials,
                              float* __restrict__ out) {
    __shared__ float red[256];
    float s = 0.0f;
    for (int i = threadIdx.x; i < NUM_WAVES; i += 256) s += partials[i];
    red[threadIdx.x] = s;
    __syncthreads();
    for (int stride = 128; stride > 0; stride >>= 1) {
        if ((int)threadIdx.x < stride) red[threadIdx.x] += red[threadIdx.x + stride];
        __syncthreads();
    }
    if (threadIdx.x == 0) out[0] = -red[0] / (float)N_SAMPLES;
}

// ---------------------------------------------------------------------------
extern "C" void kernel_launch(void* const* d_in, const int* in_sizes, int n_in,
                              void* d_out, int out_size, void* d_ws, size_t ws_size,
                              hipStream_t stream) {
    const int*   Y     = (const int*)d_in[0];     // [65536, 512] int32
    const float* theta = (const float*)d_in[1];   // [512] f32
    const float* psi   = (const float*)d_in[2];   // [512] f32

    float* ws       = (float*)d_ws;               // [0..511]=u, [512..1023]=v
    float* partials = ws + 2 * NUM_F;             // [1024] wave partials
    float* out      = (float*)d_out;

    mnar_init<<<1, 512, 0, stream>>>(theta, psi, ws);
    mnar_count<<<NUM_BLOCKS, 256, 0, stream>>>(Y, ws, partials);
    mnar_finalize<<<1, 256, 0, stream>>>(partials, out);
}
